// BiLstmCRFNer_51573967290719
// MI455X (gfx1250) — matrix operand order repaired and checked by reading, not compiled, not run on hardware
//
#include <hip/hip_runtime.h>

// ---------------- problem constants ----------------
#define B_  32
#define T_  256
#define E_  256
#define H_  512
#define G_  2048          // 4*H
#define K_  9
#define MB_ 8192          // T_*B_

// ---------------- WMMA plumbing ----------------
typedef __attribute__((ext_vector_type(4)))  unsigned int v4u;
typedef __attribute__((ext_vector_type(8)))  float        v8f;
typedef __attribute__((ext_vector_type(16))) __bf16       v16bf;

union FragU { v4u u[2]; v16bf v; };

__device__ inline unsigned short f2bf(float f) {
  unsigned u = __float_as_uint(f);
  u += 0x7fffu + ((u >> 16) & 1u);        // round-to-nearest-even
  return (unsigned short)(u >> 16);
}

__device__ inline float sigm(float x) { return 1.f / (1.f + expf(-x)); }

// ---- CDNA5 async global->LDS copy (ASYNCcnt-tracked), 16B per lane ----
__device__ inline unsigned lds_off(const void* p) { return (unsigned)(size_t)p; }

__device__ inline void async_cp16(unsigned lds_addr, const void* g) {
  asm volatile("global_load_async_to_lds_b128 %0, %1, off"
               :: "v"(lds_addr), "v"((unsigned long long)(size_t)g)
               : "memory");
}
__device__ inline void s_wait_async() {
  asm volatile("s_wait_asynccnt 0x0" ::: "memory");
}

// A fragment 16x32 bf16 (row-major source, stride lda elements)
__device__ inline FragU load_a_frag(const unsigned short* A, int lda, int row0, int k0) {
  int lane = threadIdx.x & 31;
  int m = lane & 15, kh = lane >> 4;
  const unsigned short* p = A + (size_t)(row0 + m) * lda + k0 + kh * 16;
  FragU f;
  f.u[0] = *(const v4u*)p;          // K = k0+kh*16   ..+7  -> VGPR0-3
  f.u[1] = *(const v4u*)(p + 8);    // K = k0+kh*16+8 ..+15 -> VGPR4-7
  return f;
}

// B fragment 32x16 bf16 where B(k,n) = W[n][k], W row-major stride ldb
__device__ inline FragU load_b_frag(const unsigned short* W, int ldb, int n0, int k0) {
  int lane = threadIdx.x & 31;
  int n = lane & 15, kh = lane >> 4;
  const unsigned short* p = W + (size_t)(n0 + n) * ldb + k0 + kh * 16;
  FragU f;
  f.u[0] = *(const v4u*)p;          // K = k0+kh*16   ..+7
  f.u[1] = *(const v4u*)(p + 8);    // K = k0+kh*16+8 ..+15
  return f;
}

__device__ inline v8f wmma_bf16(FragU a, FragU b, v8f c) {
  return __builtin_amdgcn_wmma_f32_16x16x32_bf16(false, a.v, false, b.v,
                                                 (short)0, c, false, false);
}

// ---------------- global sense-reversing barrier (per direction) ----------------
__device__ inline void dir_barrier(int* bar, int nwg) {
  __syncthreads();
  if (threadIdx.x == 0) {
    volatile int* vb = (volatile int*)bar;
    int gen = vb[1];
    __threadfence();
    if (atomicAdd(&bar[0], 1) == nwg - 1) {
      bar[0] = 0;
      __threadfence();
      atomicAdd(&bar[1], 1);
    } else {
      while (vb[1] == gen) __builtin_amdgcn_s_sleep(2);
    }
  }
  __syncthreads();
}

// ---------------- kernels ----------------

// Embedding gather (B,T)->(T,B,E) bf16, masked
__global__ void k_embed(const int* __restrict__ ids, const int* __restrict__ mask,
                        const float* __restrict__ emb, unsigned short* __restrict__ x0) {
  int r = blockIdx.x;            // r = t*B + b
  int t = r >> 5, b = r & 31;
  int id = ids[b * T_ + t];
  float mf = (float)mask[b * T_ + t];
  const float* e = emb + (size_t)id * E_;
  x0[(size_t)r * E_ + threadIdx.x] = f2bf(e[threadIdx.x] * mf);
}

// f32 -> bf16 weight convert
__global__ void k_cvt(const float* __restrict__ in, unsigned short* __restrict__ out, int n) {
  int i = blockIdx.x * 256 + threadIdx.x;
  if (i < n) out[i] = f2bf(in[i]);
}

// Wc (9x1024) -> padded (16x1024) bf16
__global__ void k_wcpad(const float* __restrict__ wc, unsigned short* __restrict__ out) {
  int i = blockIdx.x * 256 + threadIdx.x;     // 16384 elems
  int n = i >> 10, k = i & 1023;
  out[i] = (n < K_) ? f2bf(wc[n * 1024 + k]) : (unsigned short)0;
}

// Input projection: C[M,2048] = A[M,Kd] @ W[2048,Kd]^T + (b1+b2).
// The 16x32 A-tile is shared by all 8 waves -> stage it once via async
// global->LDS (double-buffered); B fragments stream from L2.
__global__ __launch_bounds__(256) void k_proj(const unsigned short* __restrict__ A,
                                              const unsigned short* __restrict__ W,
                                              const float* __restrict__ b1,
                                              const float* __restrict__ b2,
                                              float* __restrict__ C, int Kd) {
  __shared__ unsigned short sA[2][16 * 32];   // two 1KB A-tile buffers
  int tid = threadIdx.x;
  int wave = tid >> 5;
  int m0 = blockIdx.x * 16;
  int n0 = blockIdx.y * 128 + wave * 16;
  int nk = Kd >> 5;
  int sm = tid >> 2, sq = tid & 3;            // staging: 64 lanes x 16B

  if (tid < 64) {
    async_cp16(lds_off(&sA[0][sm * 32 + sq * 8]),
               A + (size_t)(m0 + sm) * Kd + sq * 8);
    s_wait_async();
  }
  __syncthreads();

  v8f acc = {};
  for (int kc = 0; kc < nk; ++kc) {
    int cur = kc & 1;
    if (kc + 1 < nk && tid < 64)             // prefetch next chunk
      async_cp16(lds_off(&sA[cur ^ 1][sm * 32 + sq * 8]),
                 A + (size_t)(m0 + sm) * Kd + (kc + 1) * 32 + sq * 8);
    FragU a = load_a_frag(&sA[cur][0], 32, 0, 0);
    FragU b = load_b_frag(W, Kd, n0, kc * 32);
    acc = wmma_bf16(a, b, acc);
    if (tid < 64) s_wait_async();
    __syncthreads();
  }
  int lane = tid & 31;
  int n = n0 + (lane & 15);
  float bias = b1[n] + b2[n];
  int mh = (lane >> 4) * 8;
  for (int v = 0; v < 8; ++v)
    C[(size_t)(m0 + v + mh) * G_ + n] = acc[v] + bias;
}

// Persistent recurrent kernel: grid = 2 dirs * 32 WGs, each WG owns 16 hidden
// units. Whh slice (64 rows x 512, gate-permuted) pinned in LDS; h ping-pongs
// through L2 and is async-staged into LDS once per step (32KB shared by all
// 8 waves instead of 8x redundant global reads).
__global__ __launch_bounds__(256) void k_rec(const float* __restrict__ gin_f,
                                             const float* __restrict__ gin_b,
                                             const unsigned short* __restrict__ whh_f,
                                             const unsigned short* __restrict__ whh_b,
                                             unsigned short* __restrict__ seq,   // (T*B,1024) bf16
                                             unsigned short* __restrict__ hbuf,  // [dir][2][32*512]
                                             int* __restrict__ bar) {
  const int NWG = 32;
  int dir = blockIdx.x / NWG;
  int wg  = blockIdx.x % NWG;
  int j0  = wg * 16;
  const float*          gin = dir ? gin_b : gin_f;
  const unsigned short* whh = dir ? whh_b : whh_f;
  unsigned short* hb = hbuf + (size_t)dir * 2 * B_ * H_;
  int* mybar = bar + dir * 32;

  __shared__ unsigned short sWhh[64][520];   // 64 gate-rows x 512 (+pad), ~65 KB
  __shared__ unsigned short sH[32 * 512];    // staged h_prev, 32 KB
  __shared__ float sGate[32][68];            // 32 batch x 64 gate cols (+pad)

  int tid = threadIdx.x;

  // stage Whh slice, rows permuted: r -> gate (r>>4), hidden j0+(r&15)
  {
    int r = tid >> 2, part = tid & 3;
    int grow = (r >> 4) * H_ + j0 + (r & 15);
    const unsigned short* src = whh + (size_t)grow * H_ + part * 128;
    for (int i = 0; i < 16; ++i)
      *(v4u*)&sWhh[r][part * 128 + i * 8] = *(const v4u*)(src + i * 8);
  }
  // zero our slice of h ping buffer 0
  for (int p = tid; p < 512; p += 256) {
    int b = p >> 4, jl = p & 15;
    hb[b * H_ + j0 + jl] = 0;
  }
  __syncthreads();
  dir_barrier(mybar, NWG);

  int lane = tid & 31, wave = tid >> 5;
  int mt = wave & 1, nt = wave >> 1;         // 2 M-tiles x 4 N-tiles = 8 waves
  float c0 = 0.f, c1 = 0.f;

  for (int step = 0; step < T_; ++step) {
    int t = dir ? (T_ - 1 - step) : step;
    const unsigned short* hp = hb + (size_t)(step & 1) * B_ * H_;
    unsigned short*       hn = hb + (size_t)((step + 1) & 1) * B_ * H_;

    // async-stage h_prev (32KB linear): 256 lanes x 8 rounds x 16B
    {
      unsigned lbase = lds_off(&sH[0]) + tid * 16;
      const unsigned short* gp = hp + tid * 8;
      for (int r = 0; r < 8; ++r)
        async_cp16(lbase + r * 4096, gp + (size_t)r * 2048);
      s_wait_async();
    }
    __syncthreads();

    // (32 x 512) @ (512 x 64-slice): 16 wmma per wave, A+B both from LDS
    v8f acc = {};
    for (int k = 0; k < H_; k += 32) {
      FragU a = load_a_frag(&sH[0], H_, mt * 16, k);
      FragU b = load_b_frag(&sWhh[0][0], 520, nt * 16, k);
      acc = wmma_bf16(a, b, acc);
    }
    {
      int n = nt * 16 + (lane & 15);
      int mh = (lane >> 4) * 8;
      for (int v = 0; v < 8; ++v) sGate[mt * 16 + v + mh][n] = acc[v];
    }
    __syncthreads();

    // elementwise cell update: 512 (b,j) pairs, 2 per thread; c stays in regs
    const float* gt = gin + (size_t)t * B_ * G_;
    for (int i = 0; i < 2; ++i) {
      int p = tid + i * 256;
      int b = p >> 4, jl = p & 15;
      int jg = j0 + jl;
      const float* gb = gt + (size_t)b * G_;
      float ig = sGate[b][jl]      + gb[jg];
      float fg = sGate[b][16 + jl] + gb[H_ + jg];
      float gg = sGate[b][32 + jl] + gb[2 * H_ + jg];
      float og = sGate[b][48 + jl] + gb[3 * H_ + jg];
      float c = i ? c1 : c0;
      c = sigm(fg) * c + sigm(ig) * tanhf(gg);
      if (i) c1 = c; else c0 = c;
      float h = sigm(og) * tanhf(c);
      unsigned short hbv = f2bf(h);
      hn[b * H_ + jg] = hbv;
      seq[((size_t)t * B_ + b) * (2 * H_) + dir * H_ + jg] = hbv;
    }
    __syncthreads();
    dir_barrier(mybar, NWG);
  }
}

// feats = seq1 @ WcPad^T + bc  (single 16-wide N tile, cols 9..15 discarded)
__global__ __launch_bounds__(256) void k_feats(const unsigned short* __restrict__ seq1,
                                               const unsigned short* __restrict__ wcp,
                                               const float* __restrict__ bc,
                                               float* __restrict__ feats,
                                               float* __restrict__ outF) {
  int wave = threadIdx.x >> 5;
  int m0 = (blockIdx.x * 8 + wave) * 16;
  v8f acc = {};
  for (int k = 0; k < 2 * H_; k += 32) {
    FragU a = load_a_frag(seq1, 2 * H_, m0, k);
    FragU b = load_b_frag(wcp, 2 * H_, 0, k);
    acc = wmma_bf16(a, b, acc);
  }
  int lane = threadIdx.x & 31;
  int n = lane & 15;
  int mh = (lane >> 4) * 8;
  if (n < K_) {
    float bias = bc[n];
    for (int v = 0; v < 8; ++v) {
      int m = m0 + v + mh;                 // m = t*B + b
      int t = m >> 5, b = m & 31;
      float val = acc[v] + bias;
      feats[(size_t)m * 16 + n] = val;
      outF[1 + ((size_t)b * T_ + t) * K_ + n] = val;
    }
  }
}

// CRF: forward algorithm + gold-path score, single WG (9 waves)
__global__ __launch_bounds__(288) void k_crf(const float* __restrict__ feats,
                                             const int* __restrict__ labels,
                                             const int* __restrict__ mask,
                                             const float* __restrict__ startv,
                                             const float* __restrict__ endv,
                                             const float* __restrict__ trans,
                                             float* __restrict__ out) {
  __shared__ float sT[81];
  __shared__ float sA[32][12];
  __shared__ float sDen[32], sNum[32];
  int tid = threadIdx.x;
  if (tid < 81) sT[tid] = trans[tid];
  __syncthreads();
  int b = tid / 9, k = tid % 9;

  float alpha = startv[k] + feats[(size_t)b * 16 + k];
  for (int t = 1; t < T_; ++t) {
    sA[b][k] = alpha;
    __syncthreads();
    float mx = -1e30f;
    for (int j = 0; j < 9; ++j) mx = fmaxf(mx, sA[b][j] + sT[j * 9 + k]);
    float s = 0.f;
    for (int j = 0; j < 9; ++j) s += expf(sA[b][j] + sT[j * 9 + k] - mx);
    float nxt = mx + logf(s) + feats[(size_t)(t * B_ + b) * 16 + k];
    alpha = ((float)mask[b * T_ + t] > 0.f) ? nxt : alpha;
    __syncthreads();
  }
  sA[b][k] = alpha + endv[k];
  __syncthreads();
  if (k == 0) {
    float mx = -1e30f;
    for (int j = 0; j < 9; ++j) mx = fmaxf(mx, sA[b][j]);
    float s = 0.f;
    for (int j = 0; j < 9; ++j) s += expf(sA[b][j] - mx);
    sDen[b] = mx + logf(s);
    // gold path score
    int tg0 = labels[b * T_];
    float num = startv[tg0] + feats[(size_t)b * 16 + tg0];
    int ptg = tg0, se = mask[b * T_];
    for (int t = 1; t < T_; ++t) {
      float mf = (float)mask[b * T_ + t];
      int tg = labels[b * T_ + t];
      num += (sT[ptg * 9 + tg] + feats[(size_t)(t * B_ + b) * 16 + tg]) * mf;
      ptg = tg;
      se += mask[b * T_ + t];
    }
    num += endv[labels[b * T_ + (se - 1)]];
    sNum[b] = num;
  }
  __syncthreads();
  if (tid == 0) {
    float acc = 0.f;
    for (int i = 0; i < B_; ++i) acc += sNum[i] - sDen[i];
    out[0] = -acc;
  }
}

// ---------------- host ----------------
extern "C" void kernel_launch(void* const* d_in, const int* in_sizes, int n_in,
                              void* d_out, int out_size, void* d_ws, size_t ws_size,
                              hipStream_t stream) {
  (void)in_sizes; (void)n_in; (void)out_size; (void)ws_size;
  const int*   ids     = (const int*)d_in[0];
  const int*   mask    = (const int*)d_in[1];
  const int*   labels  = (const int*)d_in[2];
  const float* emb     = (const float*)d_in[3];
  const float* l0Wihf  = (const float*)d_in[4];
  const float* l0Whhf  = (const float*)d_in[5];
  const float* l0bihf  = (const float*)d_in[6];
  const float* l0bhhf  = (const float*)d_in[7];
  const float* l0Wihb  = (const float*)d_in[8];
  const float* l0Whhb  = (const float*)d_in[9];
  const float* l0bihb  = (const float*)d_in[10];
  const float* l0bhhb  = (const float*)d_in[11];
  const float* l1Wihf  = (const float*)d_in[12];
  const float* l1Whhf  = (const float*)d_in[13];
  const float* l1bihf  = (const float*)d_in[14];
  const float* l1bhhf  = (const float*)d_in[15];
  const float* l1Wihb  = (const float*)d_in[16];
  const float* l1Whhb  = (const float*)d_in[17];
  const float* l1bihb  = (const float*)d_in[18];
  const float* l1bhhb  = (const float*)d_in[19];
  const float* Wc      = (const float*)d_in[20];
  const float* bc      = (const float*)d_in[21];
  const float* crfS    = (const float*)d_in[22];
  const float* crfE    = (const float*)d_in[23];
  const float* crfT    = (const float*)d_in[24];
  float* out = (float*)d_out;

  char* ws = (char*)d_ws;
  size_t off = 0;
  auto alloc = [&](size_t bytes) -> void* {
    void* p = (void*)(ws + off);
    off += (bytes + 4095) & ~((size_t)4095);
    return p;
  };
  int*            bar  = (int*)           alloc(4096);
  unsigned short* x0   = (unsigned short*)alloc((size_t)MB_ * E_ * 2);
  unsigned short* w0f  = (unsigned short*)alloc((size_t)G_ * E_ * 2);
  unsigned short* w0b  = (unsigned short*)alloc((size_t)G_ * E_ * 2);
  unsigned short* hh0f = (unsigned short*)alloc((size_t)G_ * H_ * 2);
  unsigned short* hh0b = (unsigned short*)alloc((size_t)G_ * H_ * 2);
  unsigned short* hh1f = (unsigned short*)alloc((size_t)G_ * H_ * 2);
  unsigned short* hh1b = (unsigned short*)alloc((size_t)G_ * H_ * 2);
  unsigned short* w1f  = (unsigned short*)alloc((size_t)G_ * 1024 * 2);
  unsigned short* w1b  = (unsigned short*)alloc((size_t)G_ * 1024 * 2);
  unsigned short* wcp  = (unsigned short*)alloc((size_t)16 * 1024 * 2);
  float*          ginF = (float*)         alloc((size_t)MB_ * G_ * 4);
  float*          ginB = (float*)         alloc((size_t)MB_ * G_ * 4);
  unsigned short* seq0 = (unsigned short*)alloc((size_t)MB_ * 1024 * 2);
  unsigned short* seq1 = (unsigned short*)alloc((size_t)MB_ * 1024 * 2);
  unsigned short* hbuf = (unsigned short*)alloc((size_t)2 * 2 * B_ * H_ * 2);
  float*          fts  = (float*)         alloc((size_t)MB_ * 16 * 4);

  hipMemsetAsync(bar, 0, 4096, stream);
  k_embed<<<MB_, 256, 0, stream>>>(ids, mask, emb, x0);
  k_cvt<<<(G_ * E_ + 255) / 256, 256, 0, stream>>>(l0Wihf, w0f, G_ * E_);
  k_cvt<<<(G_ * E_ + 255) / 256, 256, 0, stream>>>(l0Wihb, w0b, G_ * E_);
  k_cvt<<<(G_ * H_ + 255) / 256, 256, 0, stream>>>(l0Whhf, hh0f, G_ * H_);
  k_cvt<<<(G_ * H_ + 255) / 256, 256, 0, stream>>>(l0Whhb, hh0b, G_ * H_);
  k_cvt<<<(G_ * H_ + 255) / 256, 256, 0, stream>>>(l1Whhf, hh1f, G_ * H_);
  k_cvt<<<(G_ * H_ + 255) / 256, 256, 0, stream>>>(l1Whhb, hh1b, G_ * H_);
  k_cvt<<<(G_ * 1024 + 255) / 256, 256, 0, stream>>>(l1Wihf, w1f, G_ * 1024);
  k_cvt<<<(G_ * 1024 + 255) / 256, 256, 0, stream>>>(l1Wihb, w1b, G_ * 1024);
  k_wcpad<<<64, 256, 0, stream>>>(Wc, wcp);

  // layer 0
  k_proj<<<dim3(MB_ / 16, 16), 256, 0, stream>>>(x0, w0f, l0bihf, l0bhhf, ginF, E_);
  k_proj<<<dim3(MB_ / 16, 16), 256, 0, stream>>>(x0, w0b, l0bihb, l0bhhb, ginB, E_);
  k_rec<<<64, 256, 0, stream>>>(ginF, ginB, hh0f, hh0b, seq0, hbuf, bar);
  // layer 1 (reuse gin buffers)
  k_proj<<<dim3(MB_ / 16, 16), 256, 0, stream>>>(seq0, w1f, l1bihf, l1bhhf, ginF, 1024);
  k_proj<<<dim3(MB_ / 16, 16), 256, 0, stream>>>(seq0, w1b, l1bihb, l1bhhb, ginB, 1024);
  k_rec<<<64, 256, 0, stream>>>(ginF, ginB, hh1f, hh1b, seq1, hbuf, bar + 256);

  k_feats<<<64, 256, 0, stream>>>(seq1, wcp, bc, fts, out);
  k_crf<<<1, 288, 0, stream>>>(fts, labels, mask, crfS, crfE, crfT, out);
}
